// SupervisedContrastiveLoss_29111288332444
// MI455X (gfx1250) — matrix operand order repaired and checked by reading, not compiled, not run on hardware
//
#include <hip/hip_runtime.h>
#include <hip/hip_bf16.h>

// ---------------------------------------------------------------------------
// Supervised contrastive loss, fused. N=4096 samples, D=512 features.
//   sim = Xhat @ Xhat^T  (Xhat = row-normalized features, bf16, L2-resident)
//   S1[i] = sum_{j != i} exp(sim_ij / T)
//   S2[i] = sum_{j: lab_j==lab_i, j!=i} exp(sim_ij / T)
//   P[i]  = hist[lab_i] - 1
//   loss  = -(1/N^2) * [ (sum_i -log S1_i) * (sum_j 1/P_j)
//                        + N * sum_j log(S2_j + (N-P_j) + 1e-9)/P_j ]
// GEMM runs on v_wmma_f32_16x16x32_bf16 with two independent accumulator
// chains (even/odd k) so back-to-back WMMAs have no D->C RAW hazard.
// ---------------------------------------------------------------------------

typedef __bf16 v16bf __attribute__((ext_vector_type(16)));
typedef __bf16 v8bf  __attribute__((ext_vector_type(8)));
typedef float  v8f   __attribute__((ext_vector_type(8)));

static constexpr int   NROWS   = 4096;
static constexpr int   DIM     = 512;
static constexpr int   KTILES  = DIM / 32;               // 16 wmma k-steps
static constexpr int   NCHUNKS = 8;                      // column chunks
static constexpr int   JTILES  = (NROWS / 16) / NCHUNKS; // 32 column tiles/chunk
static constexpr float INV_T_LOG2E = 14.426950408889634f; // log2(e)/0.1

__device__ __forceinline__ float fast_exp2(float x) {
#if __has_builtin(__builtin_amdgcn_exp2f)
  return __builtin_amdgcn_exp2f(x);   // raw v_exp_f32; args bounded ~[-14.5,14.5]
#else
  return exp2f(x);
#endif
}

// ---------------------------------------------------------------------------
// Kernel 0: zero the accumulators (ws is poisoned; must re-zero every call)
// ---------------------------------------------------------------------------
__global__ void zero_ws_kernel(float* __restrict__ s1, float* __restrict__ s2,
                               int* __restrict__ hist) {
  int i = blockIdx.x * blockDim.x + threadIdx.x;
  if (i < NROWS) { s1[i] = 0.0f; s2[i] = 0.0f; }
  if (i < 16)    hist[i] = 0;
}

// ---------------------------------------------------------------------------
// Kernel 1: row L2-normalize -> bf16, plus label histogram.
// One 128-thread block per row; folds the cosine denominator exactly.
// ---------------------------------------------------------------------------
__global__ void __launch_bounds__(128)
normalize_kernel(const float* __restrict__ X, const int* __restrict__ labels,
                 __bf16* __restrict__ Xb, int* __restrict__ hist) {
  const int row = blockIdx.x;
  const int t   = threadIdx.x;
  const float* xr = X + (size_t)row * DIM;

  float v[4];
  float ss = 0.0f;
#pragma unroll
  for (int k = 0; k < 4; ++k) { v[k] = xr[t + 128 * k]; ss += v[k] * v[k]; }

  __shared__ float red[128];
  red[t] = ss;
  __syncthreads();
#pragma unroll
  for (int s = 64; s > 0; s >>= 1) {
    if (t < s) red[t] += red[t + s];
    __syncthreads();
  }
  const float inv = rsqrtf(red[0] + 1e-24f);

  __bf16* out = Xb + (size_t)row * DIM;
#pragma unroll
  for (int k = 0; k < 4; ++k) out[t + 128 * k] = (__bf16)(v[k] * inv);

  if (t == 0) atomicAdd(&hist[labels[row] & 15], 1);
}

// ---------------------------------------------------------------------------
// Kernel 2: fused WMMA GEMM + exp + masked row reductions.
// One wave (32 threads) per block. blockIdx.x = M-tile (16 rows),
// blockIdx.y = 512-column N-chunk. A fragments for all K held in VGPRs.
//
// bf16 A 16x32 layout: lane l -> row (l&15); K in [16*(l>>4), +8) at elems 0..7
//                      and [16*(l>>4)+8, +8) at elems 8..15 (contiguous 16B each).
// bf16 B 32x16 layout: lane l -> col (l&15); K in [16*(l>>4), +16) contiguous 32B.
// f32  C 16x16 layout: lane l -> col (l&15); VGPR r -> row 8*(l>>4)+r.
// Since B is a transposed X tile, both fragments are contiguous X row segments.
// ---------------------------------------------------------------------------
__global__ void __launch_bounds__(32)
supcon_wmma_kernel(const __bf16* __restrict__ Xb, const int* __restrict__ labels,
                   float* __restrict__ S1, float* __restrict__ S2) {
  const int mt    = blockIdx.x;
  const int chunk = blockIdx.y;
  const int lane  = threadIdx.x;
  const int half  = lane >> 4;
  const int lm    = lane & 15;
  const int mbase = mt * 16;

  // ---- A fragments: 16 rows x 512 K, resident in 128 VGPRs -----------------
  const __bf16* arow = Xb + (size_t)(mbase + lm) * DIM;
  const int kb = half * 16;
  v16bf afrag[KTILES];
#pragma unroll
  for (int kt = 0; kt < KTILES; ++kt) {
    const __bf16* p = arow + kt * 32;
    v8bf lo = *(const v8bf*)(p + kb);        // K = kb .. kb+7
    v8bf hi = *(const v8bf*)(p + kb + 8);    // K = kb+8 .. kb+15
    v16bf a;
#pragma unroll
    for (int i = 0; i < 8; ++i) { a[i] = lo[i]; a[i + 8] = hi[i]; }
    afrag[kt] = a;
  }

  // labels of the 8 rows this half-wave's C registers cover
  int labm[8];
#pragma unroll
  for (int r = 0; r < 8; ++r) labm[r] = labels[mbase + half * 8 + r];

  float acc1[8], acc2[8];
#pragma unroll
  for (int r = 0; r < 8; ++r) { acc1[r] = 0.0f; acc2[r] = 0.0f; }

  const int kb2 = half * 16;

  for (int jt = 0; jt < JTILES; ++jt) {
    const int jbase = chunk * (16 * JTILES) + jt * 16;
    const int jcol  = jbase + lm;
    const __bf16* bp = Xb + (size_t)jcol * DIM + kb2;
    const int labj = labels[jcol];

    if (jt + 1 < JTILES)  // speculative prefetch of next B tile row
      __builtin_prefetch(Xb + (size_t)(jcol + 16) * DIM + kb2, 0, 1);

    // Two independent accumulator chains: even k-tiles -> c0, odd -> c1.
    // Adjacent WMMAs share no D/C registers, so the matrix pipe stays busy
    // within a single wave instead of stalling on the accumulator RAW.
    v8f c0 = {};
    v8f c1 = {};
#pragma unroll
    for (int kt = 0; kt < KTILES; kt += 2) {
      v16bf b0 = *(const v16bf*)(bp + kt * 32);        // contiguous 32B
      v16bf b1 = *(const v16bf*)(bp + (kt + 1) * 32);  // contiguous 32B
      c0 = __builtin_amdgcn_wmma_f32_16x16x32_bf16(
               false, afrag[kt], false, b0, (short)0, c0, false, false);
      c1 = __builtin_amdgcn_wmma_f32_16x16x32_bf16(
               false, afrag[kt + 1], false, b1, (short)0, c1, false, false);
    }

#pragma unroll
    for (int r = 0; r < 8; ++r) {
      const int gm = mbase + half * 8 + r;        // global row of c[r]
      const float s = c0[r] + c1[r];              // merge the two chains
      float e = fast_exp2(s * INV_T_LOG2E);       // exp(sim/T)
      if (gm == jcol) e = 0.0f;                   // diagonal: exp(-1e10) -> 0
      acc1[r] += e;
      acc2[r] += (labj == labm[r]) ? e : 0.0f;
    }
  }

  // ---- reduce across the 16 lanes holding the same rows, then commit -------
#pragma unroll
  for (int r = 0; r < 8; ++r) {
    float a1 = acc1[r], a2 = acc2[r];
#pragma unroll
    for (int m = 8; m >= 1; m >>= 1) {
      a1 += __shfl_xor(a1, m, 16);
      a2 += __shfl_xor(a2, m, 16);
    }
    if (lm == 0) {
      const int gm = mbase + half * 8 + r;
      atomicAdd(&S1[gm], a1);
      atomicAdd(&S2[gm], a2);
    }
  }
}

// ---------------------------------------------------------------------------
// Kernel 3: fold S1/S2/P into the scalar loss (single block).
// ---------------------------------------------------------------------------
__global__ void __launch_bounds__(256)
finalize_kernel(const float* __restrict__ S1, const float* __restrict__ S2,
                const int* __restrict__ labels, const int* __restrict__ hist,
                float* __restrict__ out) {
  const int t = threadIdx.x;
  float sumA = 0.0f, sumInvP = 0.0f, sumBP = 0.0f;
  for (int i = t; i < NROWS; i += 256) {
    const int   p  = hist[labels[i] & 15] - 1;
    const float fp = (float)p;
    const float A  = -logf(S1[i]);
    const float B  = logf(S2[i] + (float)(NROWS - p) + 1e-9f);
    sumA    += A;
    sumInvP += 1.0f / fp;
    sumBP   += B / fp;
  }
  __shared__ float rA[256], rI[256], rB[256];
  rA[t] = sumA; rI[t] = sumInvP; rB[t] = sumBP;
  __syncthreads();
#pragma unroll
  for (int s = 128; s > 0; s >>= 1) {
    if (t < s) { rA[t] += rA[t + s]; rI[t] += rI[t + s]; rB[t] += rB[t + s]; }
    __syncthreads();
  }
  if (t == 0) {
    const float n = (float)NROWS;
    out[0] = -(rI[0] * rA[0] + n * rB[0]) / (n * n);
  }
}

// ---------------------------------------------------------------------------
extern "C" void kernel_launch(void* const* d_in, const int* in_sizes, int n_in,
                              void* d_out, int out_size, void* d_ws, size_t ws_size,
                              hipStream_t stream) {
  (void)in_sizes; (void)n_in; (void)out_size; (void)ws_size;

  const float* X      = (const float*)d_in[0];
  const int*   labels = (const int*)d_in[1];
  float*       out    = (float*)d_out;

  char*   ws   = (char*)d_ws;
  __bf16* Xb   = (__bf16*)ws;                                  // 4 MB
  float*  S1   = (float*)(ws + (size_t)NROWS * DIM * 2);       // 16 KB
  float*  S2   = S1 + NROWS;                                   // 16 KB
  int*    hist = (int*)(S2 + NROWS);                           // 64 B

  zero_ws_kernel<<<(NROWS + 255) / 256, 256, 0, stream>>>(S1, S2, hist);
  normalize_kernel<<<NROWS, 128, 0, stream>>>(X, labels, Xb, hist);

  dim3 grid(NROWS / 16, NCHUNKS);  // 256 M-tiles x 8 column chunks = 2048 waves
  supcon_wmma_kernel<<<grid, 32, 0, stream>>>(Xb, labels, S1, S2);

  finalize_kernel<<<1, 256, 0, stream>>>(S1, S2, labels, hist, out);
}
